// Attention_79577154060295
// MI455X (gfx1250) — matrix operand order repaired and checked by reading, not compiled, not run on hardware
//
#include <hip/hip_runtime.h>
#include <stdint.h>

// ---------------- CDNA5 (gfx1250) attention: bf16 WMMA pipeline ----------------

typedef __attribute__((ext_vector_type(16))) __bf16 v16bf;
typedef __attribute__((ext_vector_type(8)))  float  v8f;
typedef __attribute__((ext_vector_type(4)))  int    v4i;

#define BATCH 4
#define SEQ   1024
#define CDIM  768
#define HEADS 12
#define HDIM  64
#define SCALE 0.125f   // 64^-0.5 (folded into qh projection epilogue)

#if defined(__has_builtin)
#if __has_builtin(__builtin_amdgcn_global_load_async_to_lds_b32) && \
    __has_builtin(__builtin_amdgcn_s_wait_asynccnt)
#define HAVE_ASYNC 1
#endif
#if __has_builtin(__builtin_amdgcn_global_load_async_to_lds_b128)
#define HAVE_ASYNC128 1
#endif
#if __has_builtin(__builtin_amdgcn_update_dpp)
#define HAVE_DPP 1
#endif
#if __has_builtin(__builtin_amdgcn_cvt_pk_bf16_f32)
#define HAVE_CVTPK 1
#endif
#endif
#ifndef HAVE_ASYNC
#define HAVE_ASYNC 0
#endif
#ifndef HAVE_ASYNC128
#define HAVE_ASYNC128 0
#endif
#ifndef HAVE_DPP
#define HAVE_DPP 0
#endif
#ifndef HAVE_CVTPK
#define HAVE_CVTPK 0
#endif

#define GAS __attribute__((address_space(1)))
#define LAS __attribute__((address_space(3)))

union AF { v16bf bf; uint32_t u[8]; };
union CF { v8f f; float x[8]; };

struct __attribute__((aligned(16))) F4 { float v[4]; };

__device__ __forceinline__ uint16_t f2bfu(float f) {
  uint32_t u = __builtin_bit_cast(uint32_t, f);
  u += 0x7FFFu + ((u >> 16) & 1u);          // round-to-nearest-even
  return (uint16_t)(u >> 16);
}

// two f32 -> packed bf16x2 in one v_cvt_pk_bf16_f32 when available
__device__ __forceinline__ uint32_t pk2bf(float a, float b) {
#if HAVE_CVTPK
  return __builtin_bit_cast(uint32_t, __builtin_amdgcn_cvt_pk_bf16_f32(a, b));
#else
  return (uint32_t)f2bfu(a) | ((uint32_t)f2bfu(b) << 16);
#endif
}

// CDNA5 ISA 7.12.2: 16-bit A-matrix 16x32 K offsets (per half-wave)
__device__ __forceinline__ int kA(int i, int half) {
  return ((i < 4) ? 2 * i : 2 * i + 8) + 8 * half;
}
// B-matrix 32x16: same K offsets as the A-matrix (per half-wave)
__device__ __forceinline__ int kB(int i, int half) {
  return ((i < 4) ? 2 * i : 2 * i + 8) + 8 * half;
}

__device__ __forceinline__ v8f wmma_bf16(const AF& a, const AF& b, v8f c) {
  return __builtin_amdgcn_wmma_f32_16x16x32_bf16(
      false, a.bf, false, b.bf, (short)0, c, false, false);
}

// ---- 16-lane row reductions on the VALU pipe (DPP ROW_XMASK butterflies) ----
#if HAVE_DPP
template <int CTRL>
__device__ __forceinline__ float dppf(float x) {
  return __builtin_bit_cast(
      float, __builtin_amdgcn_update_dpp(0, __builtin_bit_cast(int, x), CTRL,
                                         0xF, 0xF, true));
}
#endif
__device__ __forceinline__ float rowmax16(float x) {
#if HAVE_DPP
  x = fmaxf(x, dppf<0x161>(x));   // ROW_XMASK:1
  x = fmaxf(x, dppf<0x162>(x));   // ROW_XMASK:2
  x = fmaxf(x, dppf<0x164>(x));   // ROW_XMASK:4
  x = fmaxf(x, dppf<0x168>(x));   // ROW_XMASK:8
#else
  for (int w = 1; w < 16; w <<= 1) x = fmaxf(x, __shfl_xor(x, w, 32));
#endif
  return x;
}
__device__ __forceinline__ float rowsum16(float x) {
#if HAVE_DPP
  x += dppf<0x161>(x);
  x += dppf<0x162>(x);
  x += dppf<0x164>(x);
  x += dppf<0x168>(x);
#else
  for (int w = 1; w < 16; w <<= 1) x += __shfl_xor(x, w, 32);
#endif
  return x;
}

// Load A fragment from an LDS tile stored row-major [row][32] (bf16)
__device__ __forceinline__ AF ldfragA(const uint16_t* base, int row, int half) {
  AF a;
#pragma unroll
  for (int i = 0; i < 8; ++i)
    a.u[i] = *(const uint32_t*)(base + row * 32 + kA(i, half));
  return a;
}
// Load B fragment from an LDS tile stored [col][32] (K contiguous per col)
__device__ __forceinline__ AF ldfragB(const uint16_t* base, int col, int half) {
  AF a;
#pragma unroll
  for (int i = 0; i < 8; ++i)
    a.u[i] = *(const uint32_t*)(base + col * 32 + kB(i, half));
  return a;
}

// ---------------------------------------------------------------------------
// GEMM: [4096 x 768] @ [768 x 768], double-buffered LDS, 1 barrier/iter.
// AF32: A is f32 (converted to bf16 while staging).
// EPI 0: write bf16*escale in per-head [B,H,N,D] layout.  EPI 1: f32+bias.
// Block 256 = 8 waves (4 row-groups x 2 col-groups); tile 128x128, BK=32.
// ---------------------------------------------------------------------------
template <bool AF32, int EPI>
__global__ __launch_bounds__(256) void gemm768(const void* __restrict__ Ap,
                                               const float* __restrict__ W,
                                               const float* __restrict__ bias,
                                               void* __restrict__ Out,
                                               float escale) {
  __shared__ uint16_t Asm[2][128 * 32];
  __shared__ uint16_t Bsm[2][128 * 32];

  const int tid = threadIdx.x;
  const int lane = tid & 31, wave = tid >> 5;
  const int half = lane >> 4, r = lane & 15;
  const int wrow = wave & 3, wcol = wave >> 2;
  const int row0 = blockIdx.x * 128;
  const int col0 = blockIdx.y * 128;

  F4 av[4], bv[4];
  uint64_t a2[4];

  auto loadA = [&](int kk0) {
#pragma unroll
    for (int s = 0; s < 4; ++s) {
      int flat = (tid + s * 256) * 4;
      int rr = flat >> 5, k = flat & 31;
      if (AF32) {
        const float* A = (const float*)Ap;
        av[s] = *(const F4*)(A + (size_t)(row0 + rr) * CDIM + kk0 + k);
      } else {
        const uint16_t* A = (const uint16_t*)Ap;
        a2[s] = *(const uint64_t*)(A + (size_t)(row0 + rr) * CDIM + kk0 + k);
      }
    }
  };
  auto loadB = [&](int kk0) {
#pragma unroll
    for (int s = 0; s < 4; ++s) {
      int flat = (tid + s * 256) * 4;
      int k = flat >> 7, cc = flat & 127;
      bv[s] = *(const F4*)(W + (size_t)(kk0 + k) * CDIM + col0 + cc);
    }
  };
  auto storeTiles = [&](int buf) {
#pragma unroll
    for (int s = 0; s < 4; ++s) {
      int flat = (tid + s * 256) * 4;
      int rr = flat >> 5, ka = flat & 31;
      uint64_t pk;
      if (AF32) {
        uint32_t lo = pk2bf(av[s].v[0], av[s].v[1]);
        uint32_t hi = pk2bf(av[s].v[2], av[s].v[3]);
        pk = (uint64_t)lo | ((uint64_t)hi << 32);
      } else {
        pk = a2[s];
      }
      *(uint64_t*)(&Asm[buf][rr * 32 + ka]) = pk;
      int kb = flat >> 7, cc = flat & 127;
      uint32_t b01 = pk2bf(bv[s].v[0], bv[s].v[1]);
      uint32_t b23 = pk2bf(bv[s].v[2], bv[s].v[3]);
      Bsm[buf][(cc + 0) * 32 + kb] = (uint16_t)b01;
      Bsm[buf][(cc + 1) * 32 + kb] = (uint16_t)(b01 >> 16);
      Bsm[buf][(cc + 2) * 32 + kb] = (uint16_t)b23;
      Bsm[buf][(cc + 3) * 32 + kb] = (uint16_t)(b23 >> 16);
    }
  };

  CF acc[2][4];
#pragma unroll
  for (int i = 0; i < 2; ++i)
#pragma unroll
    for (int j = 0; j < 4; ++j)
#pragma unroll
      for (int e = 0; e < 8; ++e) acc[i][j].x[e] = 0.f;

  // prologue: stage tile 0
  loadA(0);
  loadB(0);
  storeTiles(0);

  const int KT = CDIM / 32;  // 24
  int cur = 0;
  for (int kt = 0; kt < KT; ++kt) {
    __syncthreads();  // tile 'cur' visible to all; prev reads of 'cur^1' done
    if (kt + 1 < KT) {
      loadA((kt + 1) * 32);
      loadB((kt + 1) * 32);
    }
    AF a[2], b[4];
#pragma unroll
    for (int i = 0; i < 2; ++i)
      a[i] = ldfragA(Asm[cur], wrow * 32 + i * 16 + r, half);
#pragma unroll
    for (int j = 0; j < 4; ++j)
      b[j] = ldfragB(Bsm[cur], wcol * 64 + j * 16 + r, half);
#pragma unroll
    for (int i = 0; i < 2; ++i)
#pragma unroll
      for (int j = 0; j < 4; ++j)
        acc[i][j].f = wmma_bf16(a[i], b[j], acc[i][j].f);
    if (kt + 1 < KT) storeTiles(cur ^ 1);
    cur ^= 1;
  }

#pragma unroll
  for (int i = 0; i < 2; ++i)
#pragma unroll
    for (int j = 0; j < 4; j += (EPI == 0 ? 2 : 1))
#pragma unroll
      for (int e = 0; e < 8; ++e) {
        int R = row0 + wrow * 32 + i * 16 + e + 8 * half;
        int Cc = col0 + wcol * 64 + j * 16 + r;
        if (EPI == 0) {
          uint16_t* O = (uint16_t*)Out;
          int bb = R >> 10, n = R & 1023, h = Cc >> 6, d = Cc & 63;
          uint32_t pk = pk2bf(acc[i][j].x[e] * escale,
                              acc[i][j + 1].x[e] * escale);
          size_t base = (((size_t)bb * HEADS + h) * SEQ + n) * HDIM;
          O[base + d] = (uint16_t)pk;
          O[base + d + 16] = (uint16_t)(pk >> 16);
        } else {
          float* O = (float*)Out;
          O[(size_t)R * CDIM + Cc] = acc[i][j].x[e] + bias[Cc];
        }
      }
}

// ---------------------------------------------------------------------------
// Fused flash attention, post-softmax multiplicative mask.
// grid = (B*H = 48, N/128 = 8), block 256.  Each wave owns 16 q-rows.
// Double-buffered K (async-to-LDS) and V (vectorized + deferred store),
// one barrier per 32-col m-step.  8 WMMA per step per wave.
// ---------------------------------------------------------------------------
__global__ __launch_bounds__(256) void attn_kernel(
    const uint16_t* __restrict__ qh, const uint16_t* __restrict__ kh,
    const uint16_t* __restrict__ vh, const float* __restrict__ mask,
    uint16_t* __restrict__ xb) {
  __shared__ uint16_t Kt[2][32 * 64];    // [m][d]
  __shared__ uint16_t Vt[2][64 * 32];    // [d][m] (transposed)
  __shared__ uint16_t Pst[8][16 * 32];   // per-wave P staging [row][m]

  const int tid = threadIdx.x, lane = tid & 31, wave = tid >> 5;
  const int half = lane >> 4, r = lane & 15;
  const int bh = blockIdx.x;
  const int b = bh / HEADS, h = bh % HEADS;
  const int n0 = blockIdx.y * 128 + wave * 16;

  const uint16_t* kbh = kh + (size_t)bh * SEQ * HDIM;
  const uint16_t* vbh = vh + (size_t)bh * SEQ * HDIM;

  // K tile: 32x64 bf16 = 4KB; async copy straight into LDS
  auto issueK = [&](int buf, int m0) {
#if HAVE_ASYNC128
    __builtin_amdgcn_global_load_async_to_lds_b128(
        (GAS v4i*)(kbh + (size_t)m0 * HDIM + tid * 8),
        (LAS v4i*)(&Kt[buf][tid * 8]), 0, 0);
#elif HAVE_ASYNC
#pragma unroll
    for (int j = 0; j < 4; ++j) {
      int dw = tid + j * 256;
      __builtin_amdgcn_global_load_async_to_lds_b32(
          (GAS int*)(kbh + (size_t)m0 * HDIM + dw * 2),
          (LAS int*)(&Kt[buf][dw * 2]), 0, 0);
    }
#else
    for (int idx = tid; idx < 32 * 64; idx += 256)
      Kt[buf][idx] = kbh[(size_t)m0 * HDIM + idx];
#endif
  };

  uint64_t vreg[2];
  auto loadV = [&](int m0) {
#pragma unroll
    for (int s = 0; s < 2; ++s) {
      int flat = (tid + s * 256) * 4;
      int m = flat >> 6, d = flat & 63;
      vreg[s] = *(const uint64_t*)(vbh + ((size_t)m0 + m) * HDIM + d);
    }
  };
  auto storeV = [&](int buf) {
#pragma unroll
    for (int s = 0; s < 2; ++s) {
      int flat = (tid + s * 256) * 4;
      int m = flat >> 6, d = flat & 63;
#pragma unroll
      for (int j = 0; j < 4; ++j)
        Vt[buf][(d + j) * 32 + m] = (uint16_t)(vreg[s] >> (16 * j));
    }
  };

  // Q fragments (A layout), D=64 split into 2 K-chunks of 32.  SCALE is
  // already folded into qh by the projection epilogue.
  const uint16_t* qbase = qh + ((size_t)bh * SEQ + n0) * HDIM;
  AF qf[2];
#pragma unroll
  for (int c = 0; c < 2; ++c)
#pragma unroll
    for (int i = 0; i < 8; ++i)
      qf[c].u[i] = *(const uint32_t*)(qbase + r * HDIM + c * 32 + kA(i, half));

  float mrow[8], zrow[8];
  CF o[4];
#pragma unroll
  for (int e = 0; e < 8; ++e) { mrow[e] = -1e30f; zrow[e] = 0.f; }
#pragma unroll
  for (int t = 0; t < 4; ++t)
#pragma unroll
    for (int e = 0; e < 8; ++e) o[t].x[e] = 0.f;

  // prologue: stage tile 0
  issueK(0, 0);
  loadV(0);
  storeV(0);

  const int MT = SEQ / 32;  // 32
  for (int it = 0; it < MT; ++it) {
    const int m0 = it * 32;
    const int cbuf = it & 1, nbuf = cbuf ^ 1;
#if HAVE_ASYNC
    __builtin_amdgcn_s_wait_asynccnt(0);
#endif
    __syncthreads();  // current K/V landed for all waves; prev reads done
    if (it + 1 < MT) {
      issueK(nbuf, m0 + 32);
      loadV(m0 + 32);
    }

    // S = Q K^T (already scaled), two 16x16 tiles over the 32-col block
    CF s[2];
#pragma unroll
    for (int t = 0; t < 2; ++t) {
#pragma unroll
      for (int e = 0; e < 8; ++e) s[t].x[e] = 0.f;
#pragma unroll
      for (int c = 0; c < 2; ++c) {
        AF kf;
#pragma unroll
        for (int i = 0; i < 8; ++i)
          kf.u[i] = *(const uint32_t*)(&Kt[cbuf][(t * 16 + r) * HDIM + c * 32 +
                                                 kB(i, half)]);
        s[t].f = wmma_bf16(qf[c], kf, s[t].f);
      }
    }

    // row max across 32 cols (VALU DPP butterfly; C-layout row = e + 8*half)
    float tmax[8];
#pragma unroll
    for (int e = 0; e < 8; ++e)
      tmax[e] = rowmax16(fmaxf(s[0].x[e], s[1].x[e]));

    float pm0[8], pm1[8];
#pragma unroll
    for (int e = 0; e < 8; ++e) {
      float nm = fmaxf(mrow[e], tmax[e]);
      float alpha = __expf(mrow[e] - nm);
      mrow[e] = nm;
      float p0 = __expf(s[0].x[e] - nm);
      float p1 = __expf(s[1].x[e] - nm);
      // denominator: UNmasked (post-softmax mask in the reference)
      zrow[e] = zrow[e] * alpha + rowsum16(p0 + p1);
#pragma unroll
      for (int t = 0; t < 4; ++t) o[t].x[e] *= alpha;
      const float* mp = mask + ((size_t)b * SEQ + n0 + e + 8 * half) * SEQ + m0;
      pm0[e] = p0 * mp[r];  // numerator: masked
      pm1[e] = p1 * mp[16 + r];
    }

    // masked P (C-layout) -> LDS -> reload as A fragment (16x32, K = m)
    uint16_t* P = Pst[wave];
#pragma unroll
    for (int e = 0; e < 8; ++e) {
      int row = e + 8 * half;
      uint32_t pk = pk2bf(pm0[e], pm1[e]);
      P[row * 32 + r] = (uint16_t)pk;
      P[row * 32 + 16 + r] = (uint16_t)(pk >> 16);
    }
    AF pa;
#pragma unroll
    for (int i = 0; i < 8; ++i)
      pa.u[i] = *(const uint32_t*)(P + r * 32 + kA(i, half));

    // O += P * V  (4 d-tiles of 16)
#pragma unroll
    for (int t = 0; t < 4; ++t) {
      AF vf;
#pragma unroll
      for (int i = 0; i < 8; ++i)
        vf.u[i] = *(const uint32_t*)(&Vt[cbuf][(t * 16 + r) * 32 + kB(i, half)]);
      o[t].f = wmma_bf16(pa, vf, o[t].f);
    }

    if (it + 1 < MT) storeV(nbuf);  // safe: nbuf readers finished pre-barrier
  }

  // x[b, n, h*64 + d] = O / Z   (bf16, consumed by output projection GEMM)
  float rz[8];
#pragma unroll
  for (int e = 0; e < 8; ++e) rz[e] = 1.0f / zrow[e];
#pragma unroll
  for (int t = 0; t < 4; t += 2)
#pragma unroll
    for (int e = 0; e < 8; ++e) {
      int row = n0 + e + 8 * half;
      uint32_t pk = pk2bf(o[t].x[e] * rz[e], o[t + 1].x[e] * rz[e]);
      size_t base = ((size_t)b * SEQ + row) * CDIM + h * HDIM;
      xb[base + t * 16 + r] = (uint16_t)pk;
      xb[base + (t + 1) * 16 + r] = (uint16_t)(pk >> 16);
    }
}

// ---------------------------------------------------------------------------
extern "C" void kernel_launch(void* const* d_in, const int* in_sizes, int n_in,
                              void* d_out, int out_size, void* d_ws,
                              size_t ws_size, hipStream_t stream) {
  (void)in_sizes; (void)n_in; (void)out_size; (void)ws_size;
  const float* q    = (const float*)d_in[0];
  const float* k    = (const float*)d_in[1];
  const float* v    = (const float*)d_in[2];
  const float* mask = (const float*)d_in[3];
  const float* Wq   = (const float*)d_in[4];
  const float* Wk   = (const float*)d_in[5];
  const float* Wv   = (const float*)d_in[6];
  const float* Wp   = (const float*)d_in[7];
  const float* bp   = (const float*)d_in[8];

  const size_t S1 = (size_t)BATCH * HEADS * SEQ * HDIM;  // elements per tensor
  uint16_t* qh = (uint16_t*)d_ws;
  uint16_t* kh = qh + S1;
  uint16_t* vh = kh + S1;
  uint16_t* xb = vh + S1;

  dim3 ggrid(32, 6), gblk(256);
  gemm768<true, 0><<<ggrid, gblk, 0, stream>>>((const void*)q, Wq, nullptr,
                                               (void*)qh, SCALE);
  gemm768<true, 0><<<ggrid, gblk, 0, stream>>>((const void*)k, Wk, nullptr,
                                               (void*)kh, 1.0f);
  gemm768<true, 0><<<ggrid, gblk, 0, stream>>>((const void*)v, Wv, nullptr,
                                               (void*)vh, 1.0f);

  attn_kernel<<<dim3(BATCH * HEADS, SEQ / 128), 256, 0, stream>>>(qh, kh, vh,
                                                                  mask, xb);

  gemm768<false, 1><<<ggrid, gblk, 0, stream>>>((const void*)xb, Wp, bp, d_out,
                                                1.0f);
}
